// RWKV7_TimeMix_55963423867097
// MI455X (gfx1250) — matrix operand order repaired and checked by reading, not compiled, not run on hardware
//
#include <hip/hip_runtime.h>
#include <math.h>

// ---------------------------------------------------------------- constants
#define B_ 4
#define T_ 1024
#define C_ 2048
#define H_ 32
#define N_ 64
#define M_ (B_ * T_)                 // 4096 token rows
#define MC ((size_t)M_ * (size_t)C_) // 8388608 elements per (B,T,C) plane

typedef __bf16 v16bf __attribute__((ext_vector_type(16)));
typedef float  v8f   __attribute__((ext_vector_type(8)));

struct U4 { unsigned int x, y, z, w; } __attribute__((aligned(16)));

__device__ __forceinline__ unsigned short f32_to_bf16_rne(float f) {
  union { float f; unsigned int u; } cv; cv.f = f;
  unsigned int u = cv.u;
  unsigned int r = u + 0x7FFFu + ((u >> 16) & 1u);
  if ((u & 0x7F800000u) == 0x7F800000u) r = u;   // Inf/NaN passthrough
  return (unsigned short)(r >> 16);
}

// gfx1250 async global->LDS load (ASYNCcnt-tracked, no VGPR round trip).
// lds_addr: 32-bit LDS byte address; gaddr: 64-bit global byte address.
__device__ __forceinline__ void async_ld_b32(unsigned lds_addr, const void* gaddr) {
  asm volatile("global_load_async_to_lds_b32 %0, %1, off"
               :: "v"(lds_addr), "v"((unsigned long long)(size_t)gaddr)
               : "memory");
}
__device__ __forceinline__ void wait_asynccnt_6() {
  asm volatile("s_wait_asynccnt 0x6" ::: "memory");
}
// generic->LDS address: AMDGPU flat->local addrspace cast is a truncate
__device__ __forceinline__ unsigned lds_addr_of(const void* p) {
  return (unsigned)(size_t)p;
}

// ---------------------------------------------------------------- K0: weight transpose + f32->bf16
// Wt[n][k] = bf16(W[k][n]);  W is (Krows x Ncols) row-major, Wt is (Ncols x Krows)
__global__ __launch_bounds__(256) void transpose_to_bf16_kernel(
    const float* __restrict__ W, unsigned short* __restrict__ Wt,
    int Krows, int Ncols) {
  __shared__ float tile[32][33];
  const int kt = blockIdx.x * 32;
  const int nt = blockIdx.y * 32;
  const int tx = threadIdx.x;          // 0..31
  const int ty = threadIdx.y;          // 0..7
  #pragma unroll
  for (int r = 0; r < 32; r += 8)
    tile[ty + r][tx] = W[(size_t)(kt + ty + r) * Ncols + nt + tx];
  __syncthreads();
  #pragma unroll
  for (int r = 0; r < 32; r += 8)
    Wt[(size_t)(nt + ty + r) * Krows + kt + tx] = f32_to_bf16_rne(tile[tx][ty + r]);
}

// ---------------------------------------------------------------- K1: token-shift mixing -> 6 bf16 activation planes
__global__ __launch_bounds__(256) void mix_kernel(
    const float* __restrict__ x,
    const float* __restrict__ mr, const float* __restrict__ mw,
    const float* __restrict__ mk, const float* __restrict__ mv,
    const float* __restrict__ ma, const float* __restrict__ mg,
    unsigned short* __restrict__ xr, unsigned short* __restrict__ xw,
    unsigned short* __restrict__ xk, unsigned short* __restrict__ xv,
    unsigned short* __restrict__ xa, unsigned short* __restrict__ xg) {
  const size_t idx = (size_t)blockIdx.x * blockDim.x + threadIdx.x;
  const int c = (int)(idx % C_);
  const int m = (int)(idx / C_);
  const int t = m % T_;
  const float xc = x[idx];
  const float xp = (t == 0) ? 0.0f : x[idx - C_];
  const float xx = xp - xc;
  xr[idx] = f32_to_bf16_rne(xc + xx * mr[c]);
  xw[idx] = f32_to_bf16_rne(xc + xx * mw[c]);
  xk[idx] = f32_to_bf16_rne(xc + xx * mk[c]);
  xv[idx] = f32_to_bf16_rne(xc + xx * mv[c]);
  xa[idx] = f32_to_bf16_rne(xc + xx * ma[c]);
  xg[idx] = f32_to_bf16_rne(xc + xx * mg[c]);
}

// ---------------------------------------------------------------- K2: WMMA bf16 GEMM with fused epilogues
// Y(MxN) = epilogue( A(MxK,bf16) @ Wt(NxK,bf16)^T + bias )
enum { OP_F32 = 0, OP_TANH_BF16 = 1, OP_DECAY_F32 = 2,
       OP_SIGMOID_F32 = 3, OP_SIGMOID_BF16 = 4, OP_BF16 = 5 };

template <int OP>
__global__ __launch_bounds__(256) void gemm_bf16_wmma(
    const unsigned short* __restrict__ A,   // M x K row-major (bf16 bits)
    const unsigned short* __restrict__ Bt,  // N x K row-major (bf16 bits) = W^T
    const float* __restrict__ bias,         // length N or nullptr
    void* __restrict__ Yv, int Mdim, int Ndim, int Kdim) {
  const int lane = threadIdx.x & 31;
  const int wv   = threadIdx.x >> 5;                 // 8 waves / block
  const int m_base = blockIdx.x * 128 + wv * 16;     // wave's 16 rows
  const int n_base = blockIdx.y * 64;                // block's 64 cols
  const int g    = lane >> 4;                        // lane half-group
  const int l15  = lane & 15;
  const int mrow = m_base + l15;

  union FragAB { v16bf v; unsigned short h[16]; unsigned int u[8]; U4 q[2]; };
  union FragC  { v8f v; float f[8]; };

  FragC acc[4];
  #pragma unroll
  for (int f = 0; f < 4; ++f)
    #pragma unroll
    for (int e = 0; e < 8; ++e) acc[f].f[e] = 0.0f;

  const unsigned short* Arow = A + (size_t)mrow * Kdim;

  for (int k0 = 0; k0 < Kdim; k0 += 32) {
    // A fragment (16x32 bf16): lanes 0-15 hold K {0..7,16..23}, lanes 16-31 hold K {8..15,24..31}
    FragAB a;
    #pragma unroll
    for (int vv = 0; vv < 4; ++vv)
      a.u[vv] = *(const unsigned int*)(Arow + k0 + g * 8 + 2 * vv);
    #pragma unroll
    for (int vv = 0; vv < 4; ++vv)
      a.u[4 + vv] = *(const unsigned int*)(Arow + k0 + 16 + g * 8 + 2 * vv);

    #pragma unroll
    for (int f = 0; f < 4; ++f) {
      // B fragment (32x16 bf16): lane n = l15, lanes 0-15 hold K {0..7,16..23}, lanes 16-31 hold K {8..15,24..31}
      const int n = n_base + f * 16 + l15;
      const unsigned short* Brow = Bt + (size_t)n * Kdim + k0 + g * 8;
      FragAB b;
      b.q[0] = *(const U4*)(Brow);
      b.q[1] = *(const U4*)(Brow + 16);
      acc[f].v = __builtin_amdgcn_wmma_f32_16x16x32_bf16(
          false, a.v, false, b.v, (short)0, acc[f].v, false, false);
    }
  }

  // C/D layout: lane group g, VGPR v -> row m_base + g*8 + v ; col = l15 (+frag*16)
  const int mst = m_base + g * 8;
  #pragma unroll
  for (int f = 0; f < 4; ++f) {
    const int n = n_base + f * 16 + l15;
    float bi = 0.0f;
    if constexpr (OP == OP_DECAY_F32 || OP == OP_SIGMOID_F32) bi = bias[n];
    #pragma unroll
    for (int vv = 0; vv < 8; ++vv) {
      const float val = acc[f].f[vv];
      const size_t oidx = (size_t)(mst + vv) * Ndim + n;
      if constexpr (OP == OP_F32) {
        ((float*)Yv)[oidx] = val;
      } else if constexpr (OP == OP_BF16) {
        ((unsigned short*)Yv)[oidx] = f32_to_bf16_rne(val);
      } else if constexpr (OP == OP_TANH_BF16) {
        ((unsigned short*)Yv)[oidx] = f32_to_bf16_rne(tanhf(val));
      } else if constexpr (OP == OP_SIGMOID_BF16) {
        ((unsigned short*)Yv)[oidx] = f32_to_bf16_rne(1.0f / (1.0f + __expf(-val)));
      } else if constexpr (OP == OP_SIGMOID_F32) {
        const float z = bi + val;
        ((float*)Yv)[oidx] = 1.0f / (1.0f + __expf(-z));
      } else if constexpr (OP == OP_DECAY_F32) {
        const float z  = bi + val;                                    // w0 + lora
        const float sp = fmaxf(-z, 0.0f) + log1pf(__expf(-fabsf(z))); // softplus(-z)
        const float w  = -sp - 0.5f;
        ((float*)Yv)[oidx] = __expf(-__expf(w));                      // decay = exp(-exp(w))
      }
    }
  }
}

// ---------------------------------------------------------------- K3: per-head normalize kk, build avec/bvec/k2
// one wave32 per (token,head); lane handles channels 2*lane, 2*lane+1
__global__ __launch_bounds__(128) void prep_scan_kernel(
    const float* __restrict__ k, const float* __restrict__ a_sig,
    const float* __restrict__ k_k, const float* __restrict__ k_a,
    float* __restrict__ avec, float* __restrict__ bvec, float* __restrict__ k2) {
  const int hd   = blockIdx.x * 4 + (threadIdx.x >> 5); // (m*H + h)
  const int lane = threadIdx.x & 31;
  const size_t base = (size_t)hd * 64;
  const int cc = (hd % H_) * 64;            // channel base within C for params
  const int j0 = 2 * lane, j1 = j0 + 1;

  const float kk0 = k[base + j0] * k_k[cc + j0];
  const float kk1 = k[base + j1] * k_k[cc + j1];
  float ss = kk0 * kk0 + kk1 * kk1;
  #pragma unroll
  for (int off = 16; off > 0; off >>= 1) ss += __shfl_xor(ss, off, 32);
  const float inv = 1.0f / fmaxf(sqrtf(ss), 1e-12f);

  const float a0v = a_sig[base + j0];
  const float a1v = a_sig[base + j1];
  const float n0 = kk0 * inv, n1 = kk1 * inv;
  avec[base + j0] = -n0;            avec[base + j1] = -n1;
  bvec[base + j0] = n0 * a0v;       bvec[base + j1] = n1 * a1v;
  k2[base + j0] = k[base + j0] * (1.0f + (a0v - 1.0f) * k_a[cc + j0]);
  k2[base + j1] = k[base + j1] * (1.0f + (a1v - 1.0f) * k_a[cc + j1]);
}

// ---------------------------------------------------------------- K4: sequential delta-rule scan
// one 64-thread block per (b,h); thread i owns state row S[i][0..63] in VGPRs.
// Per-step operands are staged into a 2-deep LDS double buffer with gfx1250
// async global->LDS loads (ASYNCcnt), issued 2 steps ahead to hide latency.
__global__ __launch_bounds__(64) void rwkv7_scan_kernel(
    const float* __restrict__ r, const float* __restrict__ decay,
    const float* __restrict__ k2, const float* __restrict__ v,
    const float* __restrict__ avec, const float* __restrict__ bvec,
    float* __restrict__ o) {
  const int bh = blockIdx.x;          // 0 .. B*H-1
  const int b = bh / H_, h = bh % H_;
  const int i = threadIdx.x;          // 0..63

  // buf[slot][plane][j] ; planes: 0=r 1=w 2=k 3=a 4=b 5=v
  __shared__ float buf[2][6][64];

  float S[64];
  #pragma unroll
  for (int j = 0; j < 64; ++j) S[j] = 0.0f;

  const size_t base = (size_t)b * T_ * C_ + (size_t)h * 64 + i;

  // issue the 6 async loads (one b32 per lane per plane) for step t into slot
  auto issue_step = [&](int t, int slot) {
    const size_t p = base + (size_t)t * C_;
    const unsigned l = lds_addr_of(&buf[slot][0][i]);
    async_ld_b32(l +    0u, r    + p);
    async_ld_b32(l +  256u, decay+ p);
    async_ld_b32(l +  512u, k2   + p);
    async_ld_b32(l +  768u, avec + p);
    async_ld_b32(l + 1024u, bvec + p);
    async_ld_b32(l + 1280u, v    + p);
  };

  issue_step(0, 0);
  issue_step(1, 1);

  for (int t = 0; t < T_; ++t) {
    const int slot = t & 1;
    // in-order ASYNCcnt: <=6 outstanding ==> step t's 6 loads have landed
    wait_asynccnt_6();
    __syncthreads();

    const float* rs = buf[slot][0];
    const float* ws = buf[slot][1];
    const float* ks = buf[slot][2];
    const float* as = buf[slot][3];
    const float* bs = buf[slot][4];
    const float vi  = buf[slot][5][i];

    float sa = 0.0f;
    #pragma unroll
    for (int j = 0; j < 64; ++j) sa += S[j] * as[j];

    float y = 0.0f;
    #pragma unroll
    for (int j = 0; j < 64; ++j) {
      const float s = S[j] * ws[j] + sa * bs[j] + vi * ks[j];
      S[j] = s;
      y += s * rs[j];
    }
    o[base + (size_t)t * C_] = y;

    __syncthreads();                      // all lanes done reading this slot
    const int tn = (t + 2 < T_) ? (t + 2) : (T_ - 1);  // clamp keeps issue rate constant
    issue_step(tn, slot);
  }
}

// ---------------------------------------------------------------- K5: per-head epilogue (gn affine + bonus + gate) -> bf16, plus v_first out
__global__ __launch_bounds__(128) void epilogue_kernel(
    const float* __restrict__ o, const float* __restrict__ r,
    const float* __restrict__ k2, const float* __restrict__ v,
    const float* __restrict__ g, const float* __restrict__ r_k,
    const float* __restrict__ gn_scale, const float* __restrict__ gn_offset,
    unsigned short* __restrict__ og_bf, float* __restrict__ vfirst) {
  const int hd   = blockIdx.x * 4 + (threadIdx.x >> 5);
  const int lane = threadIdx.x & 31;
  const size_t base = (size_t)hd * 64;
  const int cc = (hd % H_) * 64;
  const int j0 = 2 * lane, j1 = j0 + 1;

  float dot = r[base + j0] * k2[base + j0] * r_k[cc + j0]
            + r[base + j1] * k2[base + j1] * r_k[cc + j1];
  #pragma unroll
  for (int off = 16; off > 0; off >>= 1) dot += __shfl_xor(dot, off, 32);

  #pragma unroll
  for (int q = 0; q < 2; ++q) {
    const int j = (q == 0) ? j0 : j1;
    const float vv = v[base + j];
    float ov = o[base + j] * gn_scale[cc + j] + gn_offset[cc + j];
    ov += dot * vv;
    og_bf[base + j] = f32_to_bf16_rne(ov * g[base + j]);
    vfirst[base + j] = vv;
  }
}

// ---------------------------------------------------------------- launch
extern "C" void kernel_launch(void* const* d_in, const int* in_sizes, int n_in,
                              void* d_out, int out_size, void* d_ws, size_t ws_size,
                              hipStream_t stream) {
  (void)in_sizes; (void)n_in; (void)out_size; (void)ws_size;

  const float* x     = (const float*)d_in[0];
  const float* x_r   = (const float*)d_in[1];
  const float* x_w   = (const float*)d_in[2];
  const float* x_k   = (const float*)d_in[3];
  const float* x_v   = (const float*)d_in[4];
  const float* x_a   = (const float*)d_in[5];
  const float* x_g   = (const float*)d_in[6];
  const float* w0    = (const float*)d_in[7];
  const float* w1    = (const float*)d_in[8];    // (C,64)
  const float* w2    = (const float*)d_in[9];    // (64,C)
  const float* a0    = (const float*)d_in[10];
  const float* a1    = (const float*)d_in[11];   // (C,64)
  const float* a2    = (const float*)d_in[12];   // (64,C)
  // d_in[13..15] = v0,v1,v2 : unused in first-layer path
  const float* g1    = (const float*)d_in[16];   // (C,128)
  const float* g2    = (const float*)d_in[17];   // (128,C)
  const float* k_k   = (const float*)d_in[18];
  const float* k_a   = (const float*)d_in[19];
  const float* r_k   = (const float*)d_in[20];   // (H,N) == flat C
  const float* W_r   = (const float*)d_in[21];
  const float* W_k   = (const float*)d_in[22];
  const float* W_v   = (const float*)d_in[23];
  const float* W_o   = (const float*)d_in[24];
  const float* gns   = (const float*)d_in[25];
  const float* gno   = (const float*)d_in[26];

  float* out_y   = (float*)d_out;            // (B,T,C)
  float* out_vf  = (float*)d_out + MC;       // v_first (B,T,C)

  // -------- workspace carve-out
  char* p = (char*)d_ws;
  auto alloc = [&](size_t bytes) -> void* {
    void* q = (void*)p;
    p += (bytes + 255) & ~(size_t)255;
    return q;
  };
  unsigned short* xr_bf = (unsigned short*)alloc(MC * 2);
  unsigned short* xw_bf = (unsigned short*)alloc(MC * 2);
  unsigned short* xk_bf = (unsigned short*)alloc(MC * 2);
  unsigned short* xv_bf = (unsigned short*)alloc(MC * 2);
  unsigned short* xa_bf = (unsigned short*)alloc(MC * 2);
  unsigned short* xg_bf = (unsigned short*)alloc(MC * 2);

  unsigned short* Wr_t = (unsigned short*)alloc((size_t)C_ * C_ * 2);
  unsigned short* Wk_t = (unsigned short*)alloc((size_t)C_ * C_ * 2);
  unsigned short* Wv_t = (unsigned short*)alloc((size_t)C_ * C_ * 2);
  unsigned short* Wo_t = (unsigned short*)alloc((size_t)C_ * C_ * 2);
  unsigned short* w1_t = (unsigned short*)alloc((size_t)64 * C_ * 2);
  unsigned short* w2_t = (unsigned short*)alloc((size_t)C_ * 64 * 2);
  unsigned short* a1_t = (unsigned short*)alloc((size_t)64 * C_ * 2);
  unsigned short* a2_t = (unsigned short*)alloc((size_t)C_ * 64 * 2);
  unsigned short* g1_t = (unsigned short*)alloc((size_t)128 * C_ * 2);
  unsigned short* g2_t = (unsigned short*)alloc((size_t)C_ * 128 * 2);

  float* r_f   = (float*)alloc(MC * 4);
  float* k_f   = (float*)alloc(MC * 4);
  float* v_f   = (float*)alloc(MC * 4);
  float* dec_f = (float*)alloc(MC * 4);
  float* a_f   = (float*)alloc(MC * 4);
  float* g_f   = (float*)alloc(MC * 4);
  float* av_f  = (float*)alloc(MC * 4);
  float* bv_f  = (float*)alloc(MC * 4);
  float* k2_f  = (float*)alloc(MC * 4);
  float* o_f   = (float*)alloc(MC * 4);

  unsigned short* h1_bf = (unsigned short*)alloc((size_t)M_ * 64 * 2);
  unsigned short* ha_bf = (unsigned short*)alloc((size_t)M_ * 64 * 2);
  unsigned short* hg_bf = (unsigned short*)alloc((size_t)M_ * 128 * 2);
  unsigned short* og_bf = (unsigned short*)alloc(MC * 2);

  const dim3 tb(32, 8);
  // -------- weight transpose/convert (Krows = GEMM K, Ncols = GEMM N)
  transpose_to_bf16_kernel<<<dim3(C_ / 32, C_ / 32), tb, 0, stream>>>(W_r, Wr_t, C_, C_);
  transpose_to_bf16_kernel<<<dim3(C_ / 32, C_ / 32), tb, 0, stream>>>(W_k, Wk_t, C_, C_);
  transpose_to_bf16_kernel<<<dim3(C_ / 32, C_ / 32), tb, 0, stream>>>(W_v, Wv_t, C_, C_);
  transpose_to_bf16_kernel<<<dim3(C_ / 32, C_ / 32), tb, 0, stream>>>(W_o, Wo_t, C_, C_);
  transpose_to_bf16_kernel<<<dim3(C_ / 32, 64 / 32), tb, 0, stream>>>(w1, w1_t, C_, 64);
  transpose_to_bf16_kernel<<<dim3(64 / 32, C_ / 32), tb, 0, stream>>>(w2, w2_t, 64, C_);
  transpose_to_bf16_kernel<<<dim3(C_ / 32, 64 / 32), tb, 0, stream>>>(a1, a1_t, C_, 64);
  transpose_to_bf16_kernel<<<dim3(64 / 32, C_ / 32), tb, 0, stream>>>(a2, a2_t, 64, C_);
  transpose_to_bf16_kernel<<<dim3(C_ / 32, 128 / 32), tb, 0, stream>>>(g1, g1_t, C_, 128);
  transpose_to_bf16_kernel<<<dim3(128 / 32, C_ / 32), tb, 0, stream>>>(g2, g2_t, 128, C_);

  // -------- token-shift mixing
  mix_kernel<<<(unsigned)(MC / 256), 256, 0, stream>>>(
      x, x_r, x_w, x_k, x_v, x_a, x_g,
      xr_bf, xw_bf, xk_bf, xv_bf, xa_bf, xg_bf);

  // -------- WMMA GEMMs (grid = M/128 x N/64, 256 threads)
  auto ggrid = [](int Mm, int Nn) { return dim3((unsigned)(Mm / 128), (unsigned)(Nn / 64)); };

  gemm_bf16_wmma<OP_F32><<<ggrid(M_, C_), 256, 0, stream>>>(xr_bf, Wr_t, nullptr, r_f, M_, C_, C_);
  gemm_bf16_wmma<OP_F32><<<ggrid(M_, C_), 256, 0, stream>>>(xk_bf, Wk_t, nullptr, k_f, M_, C_, C_);
  gemm_bf16_wmma<OP_F32><<<ggrid(M_, C_), 256, 0, stream>>>(xv_bf, Wv_t, nullptr, v_f, M_, C_, C_);

  gemm_bf16_wmma<OP_TANH_BF16><<<ggrid(M_, 64), 256, 0, stream>>>(xw_bf, w1_t, nullptr, h1_bf, M_, 64, C_);
  gemm_bf16_wmma<OP_DECAY_F32><<<ggrid(M_, C_), 256, 0, stream>>>(h1_bf, w2_t, w0, dec_f, M_, C_, 64);

  gemm_bf16_wmma<OP_BF16><<<ggrid(M_, 64), 256, 0, stream>>>(xa_bf, a1_t, nullptr, ha_bf, M_, 64, C_);
  gemm_bf16_wmma<OP_SIGMOID_F32><<<ggrid(M_, C_), 256, 0, stream>>>(ha_bf, a2_t, a0, a_f, M_, C_, 64);

  gemm_bf16_wmma<OP_SIGMOID_BF16><<<ggrid(M_, 128), 256, 0, stream>>>(xg_bf, g1_t, nullptr, hg_bf, M_, 128, C_);
  gemm_bf16_wmma<OP_F32><<<ggrid(M_, C_), 256, 0, stream>>>(hg_bf, g2_t, nullptr, g_f, M_, C_, 128);

  // -------- per-head kk normalization, avec/bvec/k2
  prep_scan_kernel<<<(unsigned)(M_ * H_ / 4), 128, 0, stream>>>(
      k_f, a_f, k_k, k_a, av_f, bv_f, k2_f);

  // -------- sequential delta-rule scan (128 independent (b,h) states)
  rwkv7_scan_kernel<<<B_ * H_, 64, 0, stream>>>(r_f, dec_f, k2_f, v_f, av_f, bv_f, o_f);

  // -------- per-head epilogue: gn affine + rk bonus + gate -> bf16; also v_first
  epilogue_kernel<<<(unsigned)(M_ * H_ / 4), 128, 0, stream>>>(
      o_f, r_f, k2_f, v_f, g_f, r_k, gns, gno, og_bf, out_vf);

  // -------- output projection
  gemm_bf16_wmma<OP_F32><<<ggrid(M_, C_), 256, 0, stream>>>(og_bf, Wo_t, nullptr, out_y, M_, C_, C_);
}
